// Pointnet2Backbone_10075993276693
// MI455X (gfx1250) — matrix operand order repaired and checked by reading, not compiled, not run on hardware
//
#include <hip/hip_runtime.h>
#include <hip/hip_bf16.h>
#include <stdint.h>

typedef __attribute__((ext_vector_type(16))) _Float16 v16h;
typedef __attribute__((ext_vector_type(8)))  float    v8f;
typedef _Float16 h16;

#define B_   4
#define NPTS 20000

// ---------------------------------------------------------------------------
// split pointcloud (B,N,4) -> xyz f32 (B,N,3) + feats f16 (B,N,1)
__global__ void k_split_pc(const float* __restrict__ pc, float* __restrict__ xyz,
                           h16* __restrict__ feats, int n) {
  int i = blockIdx.x * blockDim.x + threadIdx.x;
  if (i >= n) return;
  const float* p = pc + (size_t)i * 4;
  float* x = xyz + (size_t)i * 3;
  x[0] = p[0]; x[1] = p[1]; x[2] = p[2];
  feats[i] = (h16)p[3];
}

// ---------------------------------------------------------------------------
// farthest point sampling: one block per batch, min-dist kept in registers
#define FPS_BLOCK 1024
#define FPS_TPT   20   // ceil(20000/1024)
__global__ __launch_bounds__(FPS_BLOCK)
void k_fps(const float* __restrict__ xyz, int N, int npoint, int* __restrict__ inds) {
  int b = blockIdx.x, tid = threadIdx.x;
  const float* bx = xyz + (size_t)b * N * 3;
  __shared__ float smax[FPS_BLOCK];
  __shared__ int   simx[FPS_BLOCK];
  __shared__ int   s_last;
  float mind[FPS_TPT];
#pragma unroll
  for (int k = 0; k < FPS_TPT; k++) mind[k] = 1e10f;
  int last = 0;
  for (int it = 0; it < npoint; it++) {
    if (tid == 0) inds[b * npoint + it] = last;   // scan emits previous index
    float cx = bx[last * 3 + 0], cy = bx[last * 3 + 1], cz = bx[last * 3 + 2];
    float best = -1.0f; int bidx = 0;
#pragma unroll
    for (int k = 0; k < FPS_TPT; k++) {
      int p = tid + k * FPS_BLOCK;
      if (p < N) {
        float dx = bx[p * 3 + 0] - cx, dy = bx[p * 3 + 1] - cy, dz = bx[p * 3 + 2] - cz;
        float d = dx * dx + dy * dy + dz * dz;
        float m = mind[k] = fminf(mind[k], d);
        if (m > best) { best = m; bidx = p; }
      }
    }
    smax[tid] = best; simx[tid] = bidx;
    __syncthreads();
    for (int off = FPS_BLOCK / 2; off > 0; off >>= 1) {
      if (tid < off && smax[tid + off] > smax[tid]) {
        smax[tid] = smax[tid + off]; simx[tid] = simx[tid + off];
      }
      __syncthreads();
    }
    if (tid == 0) s_last = simx[0];
    __syncthreads();
    last = s_last;
  }
}

// ---------------------------------------------------------------------------
__global__ void k_gather3(const float* __restrict__ xyz, const int* __restrict__ inds,
                          float* __restrict__ nxyz, int N, int npoint) {
  int i = blockIdx.x * blockDim.x + threadIdx.x;
  if (i >= B_ * npoint) return;
  int b = i / npoint;
  int j = inds[i];
  const float* s = xyz + ((size_t)b * N + j) * 3;
  float* d = nxyz + (size_t)i * 3;
  d[0] = s[0]; d[1] = s[1]; d[2] = s[2];
}

// ---------------------------------------------------------------------------
// ball query: one wave per centroid; ordered append of first nsample hits
__global__ __launch_bounds__(32)
void k_ball_query(const float* __restrict__ xyz, const float* __restrict__ nxyz,
                  int N, int npoint, float r2, int nsample, int* __restrict__ idx) {
  int g = blockIdx.x;              // over B*npoint
  int lane = threadIdx.x;
  int b = g / npoint;
  const float* bx = xyz + (size_t)b * N * 3;
  float cx = nxyz[(size_t)g * 3 + 0], cy = nxyz[(size_t)g * 3 + 1], cz = nxyz[(size_t)g * 3 + 2];
  int* out = idx + (size_t)g * nsample;
  int count = 0, first = -1;
  for (int base = 0; base < N && count < nsample; base += 32) {
    int j = base + lane;
    bool inr = false;
    if (j < N) {
      float dx = bx[j * 3 + 0] - cx, dy = bx[j * 3 + 1] - cy, dz = bx[j * 3 + 2] - cz;
      inr = (dx * dx + dy * dy + dz * dz) < r2;
    }
    unsigned mask = (unsigned)__ballot(inr);
    if (inr) {
      int pos = count + __popc(mask & ((1u << lane) - 1u));
      if (pos < nsample) out[pos] = j;
    }
    if (first < 0 && mask) first = base + __ffs(mask) - 1;
    count += __popc(mask);
  }
  if (first < 0) first = 0;
  for (int p = count + lane; p < nsample; p += 32) out[p] = first;
}

// ---------------------------------------------------------------------------
// build grouped MLP input, K zero-padded to multiple of 32, f16
__global__ void k_group(const float* __restrict__ xyz, const h16* __restrict__ feats,
                        const float* __restrict__ nxyz, const int* __restrict__ idx,
                        h16* __restrict__ X, int N, int npoint, int nsample,
                        int C, int Kpad, float inv_r, int gc0, int nCent) {
  long long e = (long long)blockIdx.x * blockDim.x + threadIdx.x;
  long long tot = (long long)nCent * nsample * Kpad;
  if (e >= tot) return;
  int col = (int)(e % Kpad);
  long long row = e / Kpad;
  int s = (int)(row % nsample);
  int gc = gc0 + (int)(row / nsample);
  int b = gc / npoint;
  int j = idx[(size_t)gc * nsample + s];
  float v = 0.0f;
  if (col < 3)            v = (xyz[((size_t)b * N + j) * 3 + col] - nxyz[(size_t)gc * 3 + col]) * inv_r;
  else if (col < 3 + C)   v = (float)feats[((size_t)b * N + j) * C + (col - 3)];
  X[row * Kpad + col] = (h16)v;
}

// ---------------------------------------------------------------------------
__global__ void k_convert_w(const float* __restrict__ w, h16* __restrict__ wh,
                            int O, int C, int Kpad) {
  int e = blockIdx.x * blockDim.x + threadIdx.x;
  if (e >= O * Kpad) return;
  int o = e / Kpad, k = e % Kpad;
  wh[e] = (k < C) ? (h16)w[o * C + k] : (h16)0.0f;
}

// ---------------------------------------------------------------------------
// WMMA GEMM: Out(M,N) = relu(A(M,K) * W(N,K)^T + bias), f16 in, f32 acc.
// 128-thread block (4 waves) computes a 64x64 tile; each wave owns a 16-row
// strip, reuses its A fragment across 4 B fragments -> 4 WMMAs per K-step.
#define GEMM_BLOCK 128
__global__ __launch_bounds__(GEMM_BLOCK)
void k_gemm(const h16* __restrict__ A, const h16* __restrict__ W,
            const float* __restrict__ bias, h16* __restrict__ Out,
            int M, int K, int Nn) {
  __shared__ h16 lta[64 * 32];
  __shared__ h16 ltb[64 * 32];
  int tid = threadIdx.x;
  int lane = tid & 31, wave = tid >> 5;
  int tm = blockIdx.x * 64, tn = blockIdx.y * 64;
  int sr = tid >> 1, sh = tid & 1;             // staging: row 0..63, 16-col half
  const h16* ga = A + (size_t)(tm + sr) * K + sh * 16;
  const h16* gw = W + (size_t)(tn + sr) * K + sh * 16;
  int rI = lane & 15, hI = lane >> 4;          // fragment row/col + K-half select
  v8f acc[4];
#pragma unroll
  for (int j = 0; j < 4; j++)
#pragma unroll
    for (int v = 0; v < 8; v++) acc[j][v] = 0.0f;

  for (int k0 = 0; k0 < K; k0 += 32) {
    uint4 a0 = *(const uint4*)(ga + k0);
    uint4 a1 = *(const uint4*)(ga + k0 + 8);
    uint4 w0 = *(const uint4*)(gw + k0);
    uint4 w1 = *(const uint4*)(gw + k0 + 8);
    __syncthreads();
    *(uint4*)&lta[sr * 32 + sh * 16]     = a0;
    *(uint4*)&lta[sr * 32 + sh * 16 + 8] = a1;
    *(uint4*)&ltb[sr * 32 + sh * 16]     = w0;
    *(uint4*)&ltb[sr * 32 + sh * 16 + 8] = w1;
    __syncthreads();
    // A fragment for this wave's 16-row strip (16-bit A 16x32 layout:
    // lanes0-15 K=0..7/16..23, lanes16-31 K=8..15/24..31)
    v16h af;
    const h16* la = &lta[(wave * 16) * 32];
#pragma unroll
    for (int v = 0; v < 8; v++) {
      int ka = ((v < 4) ? 0 : 16) + hI * 8 + (v & 3) * 2;
      af[2 * v]     = la[rI * 32 + ka];
      af[2 * v + 1] = la[rI * 32 + ka + 1];
    }
    // 4 B fragments (16-bit B 32x16 layout: lanes0-15 K=0..7/16..23, lanes16-31 K=8..15/24..31)
#pragma unroll
    for (int j = 0; j < 4; j++) {
      v16h bf;
      const h16* lb = &ltb[(j * 16) * 32];
#pragma unroll
      for (int v = 0; v < 8; v++) {
        int kb = ((v < 4) ? 0 : 16) + hI * 8 + (v & 3) * 2;
        bf[2 * v]     = lb[rI * 32 + kb];
        bf[2 * v + 1] = lb[rI * 32 + kb + 1];
      }
      acc[j] = __builtin_amdgcn_wmma_f32_16x16x32_f16(false, af, false, bf,
                                                      (short)0, acc[j], false, false);
    }
    if (k0 + 32 < K) __builtin_prefetch(ga + k0 + 32, 0, 0);
  }
  int row0 = tm + wave * 16;
#pragma unroll
  for (int j = 0; j < 4; j++) {
    float bv = bias[tn + j * 16 + rI];
#pragma unroll
    for (int v = 0; v < 8; v++) {
      float x = fmaxf(acc[j][v] + bv, 0.0f);
      Out[(size_t)(row0 + v + 8 * hI) * Nn + (tn + j * 16 + rI)] = (h16)x;
    }
  }
}

// ---------------------------------------------------------------------------
__global__ void k_maxpool(const h16* __restrict__ A, h16* __restrict__ F,
                          int nsample, int C, int gc0, int nCent) {
  int e = blockIdx.x * blockDim.x + threadIdx.x;
  if (e >= nCent * C) return;
  int c = e % C, cl = e / C;
  const h16* a = A + (size_t)cl * nsample * C + c;
  float m = -1e30f;
  for (int s = 0; s < nsample; s++) m = fmaxf(m, (float)a[(size_t)s * C]);
  F[(size_t)(gc0 + cl) * C + c] = (h16)m;
}

// ---------------------------------------------------------------------------
__global__ void k_knn3(const float* __restrict__ x1, const float* __restrict__ x2,
                       int n1, int n2, int* __restrict__ ki, float* __restrict__ kw) {
  int p = blockIdx.x * blockDim.x + threadIdx.x;
  if (p >= B_ * n1) return;
  int b = p / n1;
  float px = x1[(size_t)p * 3], py = x1[(size_t)p * 3 + 1], pz = x1[(size_t)p * 3 + 2];
  const float* bx = x2 + (size_t)b * n2 * 3;
  float d0 = 1e30f, d1 = 1e30f, d2 = 1e30f; int i0 = 0, i1 = 0, i2 = 0;
  for (int j = 0; j < n2; j++) {
    float dx = bx[j * 3] - px, dy = bx[j * 3 + 1] - py, dz = bx[j * 3 + 2] - pz;
    float d = dx * dx + dy * dy + dz * dz;
    if (d < d0)      { d2 = d1; i2 = i1; d1 = d0; i1 = i0; d0 = d; i0 = j; }
    else if (d < d1) { d2 = d1; i2 = i1; d1 = d; i1 = j; }
    else if (d < d2) { d2 = d; i2 = j; }
  }
  float w0 = 1.0f / (d0 + 1e-8f), w1 = 1.0f / (d1 + 1e-8f), w2 = 1.0f / (d2 + 1e-8f);
  float wsum = w0 + w1 + w2;
  ki[p * 3] = i0; ki[p * 3 + 1] = i1; ki[p * 3 + 2] = i2;
  kw[p * 3] = w0 / wsum; kw[p * 3 + 1] = w1 / wsum; kw[p * 3 + 2] = w2 / wsum;
}

// ---------------------------------------------------------------------------
__global__ void k_fp_concat(const h16* __restrict__ f1, const h16* __restrict__ f2,
                            const int* __restrict__ ki, const float* __restrict__ kw,
                            h16* __restrict__ X, int n1, int n2, int C1, int C2) {
  int K = C1 + C2;
  long long e = (long long)blockIdx.x * blockDim.x + threadIdx.x;
  if (e >= (long long)B_ * n1 * K) return;
  int col = (int)(e % K);
  int p = (int)(e / K);
  int b = p / n1;
  float v;
  if (col < C1) v = (float)f1[(size_t)p * C1 + col];
  else {
    int c = col - C1;
    v = kw[p * 3 + 0] * (float)f2[((size_t)b * n2 + ki[p * 3 + 0]) * C2 + c]
      + kw[p * 3 + 1] * (float)f2[((size_t)b * n2 + ki[p * 3 + 1]) * C2 + c]
      + kw[p * 3 + 2] * (float)f2[((size_t)b * n2 + ki[p * 3 + 2]) * C2 + c];
  }
  X[(size_t)p * K + col] = (h16)v;
}

// ---------------------------------------------------------------------------
// assemble output: transpose(g2) f32, xyz2, inds1[:, :1024] as float
__global__ void k_final(const h16* __restrict__ g2, const float* __restrict__ xyz2,
                        const int* __restrict__ inds1, float* __restrict__ out) {
  const int n0 = B_ * 512 * 1024, n1 = B_ * 1024 * 3, n2 = B_ * 1024;
  int e = blockIdx.x * blockDim.x + threadIdx.x;
  if (e < n0) {
    int b = e / (512 * 1024), r = e % (512 * 1024);
    int c = r / 1024, n = r % 1024;
    out[e] = (float)g2[((size_t)b * 1024 + n) * 512 + c];
  } else if (e < n0 + n1) {
    out[e] = xyz2[e - n0];
  } else if (e < n0 + n1 + n2) {
    int i = e - n0 - n1;
    int b = i / 1024, k = i % 1024;
    out[e] = (float)inds1[b * 2048 + k];
  }
}

// ---------------------------------------------------------------------------
extern "C" void kernel_launch(void* const* d_in, const int* in_sizes, int n_in,
                              void* d_out, int out_size, void* d_ws, size_t ws_size,
                              hipStream_t stream) {
  (void)in_sizes; (void)n_in; (void)out_size; (void)ws_size;
  char* wsb = (char*)d_ws;
  size_t off = 0;
  auto alloc = [&](size_t bytes) -> void* {
    void* p = wsb + off;
    off += (bytes + 255) & ~(size_t)255;
    return p;
  };

  const float* pc = (const float*)d_in[0];
  const float* saw[4][3]; const float* sab[4][3];
  int ii = 1;
  for (int l = 0; l < 4; l++)
    for (int j = 0; j < 3; j++) { saw[l][j] = (const float*)d_in[ii++]; sab[l][j] = (const float*)d_in[ii++]; }
  const float* fpw[2][2]; const float* fpb[2][2];
  for (int l = 0; l < 2; l++)
    for (int j = 0; j < 2; j++) { fpw[l][j] = (const float*)d_in[ii++]; fpb[l][j] = (const float*)d_in[ii++]; }

  // ---- layer configs ----
  const int   np[4]     = {2048, 1024, 512, 256};
  const float rad[4]    = {0.2f, 0.4f, 0.8f, 1.2f};
  const int   nsamp[4]  = {64, 32, 16, 16};
  const int   Cin[4]    = {1, 128, 256, 256};
  const int   Kpad0[4]  = {32, 160, 288, 288};      // (3+Cin) padded to %32
  const int   Cd[4][3]  = {{64, 64, 128}, {128, 128, 256}, {128, 128, 256}, {128, 128, 256}};

  // ---- workspace ----
  float* xyz0  = (float*)alloc((size_t)B_ * NPTS * 3 * 4);
  h16*   feats0 = (h16*)alloc((size_t)B_ * NPTS * 2);
  int*   inds[4]; float* nxyz[4]; h16* feat[4];
  for (int l = 0; l < 4; l++) {
    inds[l] = (int*)alloc((size_t)B_ * np[l] * 4);
    nxyz[l] = (float*)alloc((size_t)B_ * np[l] * 3 * 4);
    feat[l] = (h16*)alloc((size_t)B_ * np[l] * Cd[l][2] * 2);
  }
  int* idxbuf = (int*)alloc((size_t)B_ * 2048 * 64 * 4);
  h16* wh[4][3];
  for (int l = 0; l < 4; l++) {
    int Ks[3] = {Kpad0[l], Cd[l][0], Cd[l][1]};
    for (int j = 0; j < 3; j++) wh[l][j] = (h16*)alloc((size_t)Cd[l][j] * Ks[j] * 2);
  }
  h16* fwh[2][2];
  const int fpK[2][2] = {{512, 512}, {768, 512}};
  for (int l = 0; l < 2; l++)
    for (int j = 0; j < 2; j++) fwh[l][j] = (h16*)alloc((size_t)512 * fpK[l][j] * 2);
  int*   ki  = (int*)alloc((size_t)B_ * 1024 * 3 * 4);
  float* kwt = (float*)alloc((size_t)B_ * 1024 * 3 * 4);
  h16*   g1buf = (h16*)alloc((size_t)B_ * 512 * 512 * 2);
  h16*   g2buf = (h16*)alloc((size_t)B_ * 1024 * 512 * 2);
  h16*   slabX = (h16*)alloc((size_t)16 << 20);
  h16*   slabA = (h16*)alloc((size_t)16 << 20);
  h16*   slabB = (h16*)alloc((size_t)16 << 20);

  // ---- preprocess ----
  k_split_pc<<<(B_ * NPTS + 255) / 256, 256, 0, stream>>>(pc, xyz0, feats0, B_ * NPTS);

  // ---- SA stages ----
  const float* curXyz = xyz0; const h16* curFeat = feats0;
  int curN = NPTS;
  for (int l = 0; l < 4; l++) {
    int Ks[3] = {Kpad0[l], Cd[l][0], Cd[l][1]};
    int Cs[3] = {3 + Cin[l], Cd[l][0], Cd[l][1]};
    for (int j = 0; j < 3; j++) {
      int ne = Cd[l][j] * Ks[j];
      k_convert_w<<<(ne + 255) / 256, 256, 0, stream>>>(saw[l][j], wh[l][j], Cd[l][j], Cs[j], Ks[j]);
    }
    k_fps<<<B_, FPS_BLOCK, 0, stream>>>(curXyz, curN, np[l], inds[l]);
    k_gather3<<<(B_ * np[l] + 255) / 256, 256, 0, stream>>>(curXyz, inds[l], nxyz[l], curN, np[l]);
    k_ball_query<<<B_ * np[l], 32, 0, stream>>>(curXyz, nxyz[l], curN, np[l],
                                                rad[l] * rad[l], nsamp[l], idxbuf);
    int totalC = B_ * np[l];
    const int chunkC = 1024;
    for (int c0 = 0; c0 < totalC; c0 += chunkC) {
      int nc = (totalC - c0 < chunkC) ? (totalC - c0) : chunkC;
      int rows = nc * nsamp[l];
      long long e0 = (long long)rows * Kpad0[l];
      k_group<<<(unsigned)((e0 + 255) / 256), 256, 0, stream>>>(
          curXyz, curFeat, nxyz[l], idxbuf, slabX, curN, np[l], nsamp[l],
          Cin[l], Kpad0[l], 1.0f / rad[l], c0, nc);
      dim3 g1((unsigned)(rows / 64), (unsigned)(Cd[l][0] / 64));
      k_gemm<<<g1, GEMM_BLOCK, 0, stream>>>(slabX, wh[l][0], sab[l][0], slabA, rows, Ks[0], Cd[l][0]);
      dim3 g2((unsigned)(rows / 64), (unsigned)(Cd[l][1] / 64));
      k_gemm<<<g2, GEMM_BLOCK, 0, stream>>>(slabA, wh[l][1], sab[l][1], slabB, rows, Ks[1], Cd[l][1]);
      dim3 g3((unsigned)(rows / 64), (unsigned)(Cd[l][2] / 64));
      k_gemm<<<g3, GEMM_BLOCK, 0, stream>>>(slabB, wh[l][2], sab[l][2], slabA, rows, Ks[2], Cd[l][2]);
      k_maxpool<<<(nc * Cd[l][2] + 255) / 256, 256, 0, stream>>>(slabA, feat[l], nsamp[l], Cd[l][2], c0, nc);
    }
    curXyz = nxyz[l]; curFeat = feat[l]; curN = np[l];
  }

  // ---- FP weights ----
  for (int l = 0; l < 2; l++)
    for (int j = 0; j < 2; j++) {
      int ne = 512 * fpK[l][j];
      k_convert_w<<<(ne + 255) / 256, 256, 0, stream>>>(fpw[l][j], fwh[l][j], 512, fpK[l][j], fpK[l][j]);
    }

  // ---- FP1: level3 (512 pts, f3 C=256) <- level4 (256 pts, f4 C=256) ----
  k_knn3<<<(B_ * 512 + 255) / 256, 256, 0, stream>>>(nxyz[2], nxyz[3], 512, 256, ki, kwt);
  {
    long long e0 = (long long)B_ * 512 * 512;
    k_fp_concat<<<(unsigned)((e0 + 255) / 256), 256, 0, stream>>>(
        feat[2], feat[3], ki, kwt, slabX, 512, 256, 256, 256);
  }
  {
    dim3 gf((unsigned)(B_ * 512 / 64), 512 / 64);
    k_gemm<<<gf, GEMM_BLOCK, 0, stream>>>(slabX, fwh[0][0], fpb[0][0], slabA, B_ * 512, 512, 512);
    k_gemm<<<gf, GEMM_BLOCK, 0, stream>>>(slabA, fwh[0][1], fpb[0][1], g1buf, B_ * 512, 512, 512);
  }

  // ---- FP2: level2 (1024 pts, f2 C=256) <- level3 (512 pts, g1 C=512) ----
  k_knn3<<<(B_ * 1024 + 255) / 256, 256, 0, stream>>>(nxyz[1], nxyz[2], 1024, 512, ki, kwt);
  {
    long long e0 = (long long)B_ * 1024 * 768;
    k_fp_concat<<<(unsigned)((e0 + 255) / 256), 256, 0, stream>>>(
        feat[1], g1buf, ki, kwt, slabX, 1024, 512, 256, 512);
  }
  {
    dim3 gf((unsigned)(B_ * 1024 / 64), 512 / 64);
    k_gemm<<<gf, GEMM_BLOCK, 0, stream>>>(slabX, fwh[1][0], fpb[1][0], slabA, B_ * 1024, 768, 512);
    k_gemm<<<gf, GEMM_BLOCK, 0, stream>>>(slabA, fwh[1][1], fpb[1][1], g2buf, B_ * 1024, 512, 512);
  }

  // ---- output ----
  int ntot = B_ * 512 * 1024 + B_ * 1024 * 3 + B_ * 1024;
  k_final<<<(ntot + 255) / 256, 256, 0, stream>>>(g2buf, nxyz[1], inds[0], (float*)d_out);
}
